// MMSingleStreamBlockRefKVOnly_15324443312833
// MI455X (gfx1250) — matrix operand order repaired and checked by reading, not compiled, not run on hardware
//
#include <hip/hip_runtime.h>
#include <hip/hip_bf16.h>

typedef __hip_bfloat16 bht;
typedef __attribute__((ext_vector_type(16))) __bf16 bfrag_t;
typedef __attribute__((ext_vector_type(8)))  float  v8f;

#define H_   3072
#define NH_  24
#define HD_  128
#define MLPD 12288
#define IMG_ 1536
#define REF_ 512
#define TXT_ 256
#define LQ   1792   // IMG+TXT (query tokens)
#define L2_  2304   // IMG+REF+TXT (kv tokens)
#define N1   21504  // 3H + MLPD
#define N2   15360  // H + MLPD
#define SCALE_QK 0.08838834764831845f

// ---------- helpers ----------
__device__ __forceinline__ bfrag_t make_frag(const bht* p0, const bht* p1) {
  union U { uint4 u[2]; bfrag_t f; } x;
  x.u[0] = *reinterpret_cast<const uint4*>(p0);
  x.u[1] = *reinterpret_cast<const uint4*>(p1);
  return x.f;
}
__device__ __forceinline__ v8f wmma_bf16(bfrag_t a, bfrag_t b, v8f c) {
  return __builtin_amdgcn_wmma_f32_16x16x32_bf16(false, a, false, b, (short)0, c, false, false);
}
__device__ __forceinline__ float gelu_tanh(float x) {
  float x3 = x * x * x;
  return 0.5f * x * (1.f + tanhf(0.7978845608028654f * (x + 0.044715f * x3)));
}
__device__ __forceinline__ float blockSum(float v) {
  __shared__ float sm[8];
  #pragma unroll
  for (int m = 16; m >= 1; m >>= 1) v += __shfl_xor(v, m, 32);
  int w = threadIdx.x >> 5;
  __syncthreads();
  if ((threadIdx.x & 31) == 0) sm[w] = v;
  __syncthreads();
  float tot = 0.f;
  int nw = blockDim.x >> 5;
  for (int i = 0; i < nw; i++) tot += sm[i];
  return tot;
}

// ---------- weight transpose + fp32->bf16 convert: w[K][N] -> wT[N][K] ----------
__global__ __launch_bounds__(256) void transpose_to_bf16(
    const float* __restrict__ w, bht* __restrict__ wT, int K, int N) {
  __shared__ float tile[32][33];
  int n0 = blockIdx.x * 32, k0 = blockIdx.y * 32;
  int tx = threadIdx.x & 31, ty = threadIdx.x >> 5;
  #pragma unroll
  for (int i = 0; i < 4; i++)
    tile[ty + 8 * i][tx] = w[(size_t)(k0 + ty + 8 * i) * N + n0 + tx];
  __syncthreads();
  #pragma unroll
  for (int i = 0; i < 4; i++)
    wT[(size_t)(n0 + ty + 8 * i) * K + k0 + tx] = __float2bfloat16(tile[tx][ty + 8 * i]);
}

__global__ __launch_bounds__(256) void f32_to_bf16(const float* __restrict__ s,
                                                   bht* __restrict__ d, size_t n4) {
  size_t i = (size_t)blockIdx.x * 256 + threadIdx.x;
  if (i >= n4) return;
  float4 v = reinterpret_cast<const float4*>(s)[i];
  d[i * 4 + 0] = __float2bfloat16(v.x);
  d[i * 4 + 1] = __float2bfloat16(v.y);
  d[i * 4 + 2] = __float2bfloat16(v.z);
  d[i * 4 + 3] = __float2bfloat16(v.w);
}

// ---------- mod = silu(vec) @ mod_w + mod_b  (1x3072 @ 3072x9216) ----------
__global__ __launch_bounds__(256) void mod_gemv(const float* __restrict__ vec,
                                                const float* __restrict__ w,
                                                const float* __restrict__ b,
                                                float* __restrict__ out) {
  __shared__ float sv[H_];
  int t = threadIdx.x;
  for (int i = t; i < H_; i += 256) {
    float x = vec[i];
    sv[i] = x / (1.f + __expf(-x));
  }
  __syncthreads();
  int n = blockIdx.x * 256 + t;
  float acc = 0.f;
  for (int k = 0; k < H_; k++) acc += sv[k] * w[(size_t)k * (3 * H_) + n];
  out[n] = acc + b[n];
}

// ---------- x_mod = LN(x)*(1+scale)+shift -> bf16 xc (row remap) ----------
__global__ __launch_bounds__(256) void ln_mod(const float* __restrict__ x,
                                              const float* __restrict__ mod,
                                              bht* __restrict__ xc) {
  int r = blockIdx.x, t = threadIdx.x;
  const float* row = x + (size_t)r * H_;
  float s = 0.f, s2 = 0.f;
  for (int i = t; i < H_; i += 256) { float v = row[i]; s += v; s2 += v * v; }
  s = blockSum(s);
  s2 = blockSum(s2);
  float mean = s / (float)H_;
  float var = s2 / (float)H_ - mean * mean;
  float inv = rsqrtf(var + 1e-6f);
  const float* shift = mod;
  const float* scale = mod + H_;
  size_t drow = (r < IMG_) ? (size_t)r : (size_t)(r + REF_);
  for (int i = t; i < H_; i += 256) {
    float v = (row[i] - mean) * inv;
    v = v * (1.f + scale[i]) + shift[i];
    xc[drow * H_ + i] = __float2bfloat16(v);
  }
}

// ---------- r = LN(rout)*g + b -> bf16 xc rows [IMG, IMG+REF) ----------
__global__ __launch_bounds__(256) void ln_ref(const float* __restrict__ src,
                                              const float* __restrict__ g,
                                              const float* __restrict__ b,
                                              bht* __restrict__ xc) {
  int r = blockIdx.x, t = threadIdx.x;
  const float* row = src + (size_t)r * H_;
  float s = 0.f, s2 = 0.f;
  for (int i = t; i < H_; i += 256) { float v = row[i]; s += v; s2 += v * v; }
  s = blockSum(s);
  s2 = blockSum(s2);
  float mean = s / (float)H_;
  float var = s2 / (float)H_ - mean * mean;
  float inv = rsqrtf(var + 1e-6f);
  size_t drow = (size_t)(IMG_ + r);
  for (int i = t; i < H_; i += 256) {
    float v = (row[i] - mean) * inv * g[i] + b[i];
    xc[drow * H_ + i] = __float2bfloat16(v);
  }
}

// ---------- generic WMMA GEMM (double-buffered LDS, 1 barrier / K-step) ----------
// C[M][N] = A[M][K](bf16) * B^T[N][K](bf16) + bias
// EPI 0: fp32 out.  EPI 1: gelu -> bf16 out.  EPI 2: resid + (acc+bias)*gate -> fp32 out.
template <int EPI>
__global__ __launch_bounds__(256) void gemm_wmma(
    const bht* __restrict__ A, const bht* __restrict__ B,
    const float* __restrict__ bias, int M, int N, int K,
    float* __restrict__ Cf, bht* __restrict__ Cb,
    const float* __restrict__ gate, const float* __restrict__ resid) {
  __shared__ __align__(16) bht Asb[2][128 * 40];
  __shared__ __align__(16) bht Bsb[2][128 * 40];
  int mBase = blockIdx.y * 128, nBase = blockIdx.x * 128;
  int t = threadIdx.x;
  int lrow = t >> 1, lseg = (t & 1) * 16;
  const bht* gA = A + (size_t)(mBase + lrow) * K + lseg;
  const bht* gB = B + (size_t)(nBase + lrow) * K + lseg;
  int sOff = lrow * 40 + lseg;

  // preload tile 0 into buffer 0
  {
    uint4 a0 = *(const uint4*)(gA);
    uint4 a1 = *(const uint4*)(gA + 8);
    uint4 b0 = *(const uint4*)(gB);
    uint4 b1 = *(const uint4*)(gB + 8);
    *(uint4*)(Asb[0] + sOff) = a0; *(uint4*)(Asb[0] + sOff + 8) = a1;
    *(uint4*)(Bsb[0] + sOff) = b0; *(uint4*)(Bsb[0] + sOff + 8) = b1;
  }

  int wave = t >> 5, lane = t & 31;
  int wm = (wave >> 2) * 64, wn = (wave & 3) * 32;
  int l16 = lane & 15;
  int kaddA = (lane < 16) ? 0 : 8;
  int kaddB = (lane < 16) ? 0 : 8;
  v8f zero = {};
  v8f acc[4][2];
  #pragma unroll
  for (int im = 0; im < 4; im++)
    #pragma unroll
    for (int in = 0; in < 2; in++) acc[im][in] = zero;

  __syncthreads();

  int nstep = K >> 5;
  int cur = 0;
  for (int i = 0; i < nstep; i++) {
    // prefetch / preload next tile while computing on current one
    uint4 a0, a1, b0, b1;
    bool more = (i + 1) < nstep;
    if (more) {
      int k0 = (i + 1) * 32;
      if (k0 + 160 <= K) {               // speculative prefetch 4 steps ahead
        __builtin_prefetch((const void*)(gA + k0 + 128), 0, 1);
        __builtin_prefetch((const void*)(gB + k0 + 128), 0, 1);
      }
      a0 = *(const uint4*)(gA + k0);
      a1 = *(const uint4*)(gA + k0 + 8);
      b0 = *(const uint4*)(gB + k0);
      b1 = *(const uint4*)(gB + k0 + 8);
    }

    const bht* As = Asb[cur];
    const bht* Bs = Bsb[cur];
    bfrag_t af[4], bfg[2];
    #pragma unroll
    for (int im = 0; im < 4; im++) {
      const bht* p = As + (wm + im * 16 + l16) * 40;
      af[im] = make_frag(p + kaddA, p + kaddA + 16);
    }
    #pragma unroll
    for (int in = 0; in < 2; in++) {
      const bht* p = Bs + (wn + in * 16 + l16) * 40;
      bfg[in] = make_frag(p + kaddB, p + kaddB + 16);
    }
    #pragma unroll
    for (int im = 0; im < 4; im++)
      #pragma unroll
      for (int in = 0; in < 2; in++)
        acc[im][in] = wmma_bf16(af[im], bfg[in], acc[im][in]);

    if (more) {
      bht* An = Asb[cur ^ 1] + sOff;
      bht* Bn = Bsb[cur ^ 1] + sOff;
      *(uint4*)(An) = a0; *(uint4*)(An + 8) = a1;
      *(uint4*)(Bn) = b0; *(uint4*)(Bn + 8) = b1;
    }
    __syncthreads();
    cur ^= 1;
  }

  int rb = (lane < 16) ? 0 : 8;
  #pragma unroll
  for (int im = 0; im < 4; im++) {
    #pragma unroll
    for (int in = 0; in < 2; in++) {
      int col = nBase + wn + in * 16 + l16;
      float bv = bias[col];
      #pragma unroll
      for (int f = 0; f < 8; f++) {
        int row = mBase + wm + im * 16 + rb + f;
        float v = acc[im][in][f] + bv;
        if (EPI == 0) {
          Cf[(size_t)row * N + col] = v;
        } else if (EPI == 1) {
          Cb[(size_t)row * N + col] = __float2bfloat16(gelu_tanh(v));
        } else {
          Cf[(size_t)row * N + col] = resid[(size_t)row * N + col] + v * gate[col];
        }
      }
    }
  }
}

// ---------- qkv prep: rms norm + rope, pack Q/K row-major and V transposed ----------
__global__ __launch_bounds__(128) void qkv_prep(
    const float* __restrict__ qkv, const float* __restrict__ cosb,
    const float* __restrict__ sinb, const float* __restrict__ qg,
    const float* __restrict__ kg, bht* __restrict__ Q, bht* __restrict__ Kd,
    bht* __restrict__ VT) {
  int b = blockIdx.x;
  int tok = b / NH_, head = b - tok * NH_;
  int d = threadIdx.x;
  const float* base = qkv + (size_t)tok * (3 * H_) + head * HD_;

  // K
  float kv = base[H_ + d];
  float ssk = blockSum(kv * kv);
  float kn = kv * rsqrtf(ssk / (float)HD_ + 1e-6f) * kg[d];
  float outk = kn;
  if (tok < IMG_) {
    float other = __shfl_xor(kn, 1, 32);
    float rot = (d & 1) ? other : -other;
    outk = kn * cosb[(size_t)tok * HD_ + d] + rot * sinb[(size_t)tok * HD_ + d];
  }
  Kd[((size_t)head * L2_ + tok) * HD_ + d] = __float2bfloat16(outk);

  // V (transposed layout [head][hd][kv])
  float vv = base[2 * H_ + d];
  VT[((size_t)head * HD_ + d) * L2_ + tok] = __float2bfloat16(vv);

  // Q (only non-ref tokens)
  int qtok = (tok < IMG_) ? tok : ((tok >= IMG_ + REF_) ? tok - REF_ : -1);
  if (qtok >= 0) {
    float qv = base[d];
    float ssq = blockSum(qv * qv);
    float qn = qv * rsqrtf(ssq / (float)HD_ + 1e-6f) * qg[d];
    float outq = qn;
    if (tok < IMG_) {
      float other = __shfl_xor(qn, 1, 32);
      float rot = (d & 1) ? other : -other;
      outq = qn * cosb[(size_t)tok * HD_ + d] + rot * sinb[(size_t)tok * HD_ + d];
    }
    Q[((size_t)head * LQ + qtok) * HD_ + d] = __float2bfloat16(outq);
  }
}

// ---------- flash attention: one wave per (head, 16-query tile) ----------
__global__ __launch_bounds__(256) void attn_wmma(
    const bht* __restrict__ Q, const bht* __restrict__ Kd,
    const bht* __restrict__ VT, const int* __restrict__ tmask,
    bht* __restrict__ out /* lin2_in, row stride N2, cols [0,H) */) {
  __shared__ unsigned char kAllow[L2_];
  __shared__ unsigned char qAllow[LQ];
  __shared__ __align__(16) bht Pb[8][16 * 32];
  int t = threadIdx.x;
  for (int i = t; i < L2_; i += 256)
    kAllow[i] = (i < IMG_ + REF_) ? 1 : (unsigned char)(tmask[i - (IMG_ + REF_)] != 0);
  for (int i = t; i < LQ; i += 256)
    qAllow[i] = (i < IMG_) ? 1 : (unsigned char)(tmask[i - IMG_] != 0);
  __syncthreads();

  int wave = t >> 5, lane = t & 31;
  int tile = blockIdx.x * 8 + wave;       // 24*112 = 2688 tiles
  int head = tile / 112, qt = tile - head * 112;
  int qBase = qt * 16;
  int l16 = lane & 15;
  int kaddA = (lane < 16) ? 0 : 8;
  int koffB = (lane < 16) ? 0 : 8;

  bfrag_t qf[4];
  const bht* qrow = Q + ((size_t)head * LQ + qBase + l16) * HD_;
  #pragma unroll
  for (int c = 0; c < 4; c++)
    qf[c] = make_frag(qrow + c * 32 + kaddA, qrow + c * 32 + kaddA + 16);

  v8f zero = {};
  v8f O[8];
  float Mrow[8], Lrow[8];
  int qg[8]; unsigned char qb[8];
  #pragma unroll
  for (int f = 0; f < 8; f++) {
    Mrow[f] = -3.0e38f; Lrow[f] = 0.f;
    qg[f] = qBase + ((lane < 16) ? f : 8 + f);
    qb[f] = qAllow[qg[f]];
  }
  #pragma unroll
  for (int d = 0; d < 8; d++) O[d] = zero;

  bht* P = Pb[wave];
  for (int s = 0; s < L2_ / 32; s++) {
    int kvB = s * 32;
    v8f s0 = zero, s1 = zero;
    #pragma unroll
    for (int c = 0; c < 4; c++) {
      const bht* kr0 = Kd + ((size_t)head * L2_ + kvB + l16) * HD_ + c * 32 + koffB;
      const bht* kr1 = kr0 + 16 * HD_;
      bfrag_t kf0 = make_frag(kr0, kr0 + 16);
      bfrag_t kf1 = make_frag(kr1, kr1 + 16);
      s0 = wmma_bf16(qf[c], kf0, s0);
      s1 = wmma_bf16(qf[c], kf1, s1);
    }
    unsigned char kb0 = kAllow[kvB + l16], kb1 = kAllow[kvB + 16 + l16];
    float p0[8], p1[8];
    #pragma unroll
    for (int f = 0; f < 8; f++) {
      float v0 = s0[f] * SCALE_QK + ((qb[f] == kb0) ? 0.f : -1e9f);
      float v1 = s1[f] * SCALE_QK + ((qb[f] == kb1) ? 0.f : -1e9f);
      float mx = fmaxf(v0, v1);
      #pragma unroll
      for (int m = 8; m >= 1; m >>= 1) mx = fmaxf(mx, __shfl_xor(mx, m, 32));
      float newM = fmaxf(Mrow[f], mx);
      float alpha = __expf(Mrow[f] - newM);
      float e0 = __expf(v0 - newM), e1 = __expf(v1 - newM);
      float rs = e0 + e1;
      #pragma unroll
      for (int m = 8; m >= 1; m >>= 1) rs += __shfl_xor(rs, m, 32);
      Mrow[f] = newM;
      Lrow[f] = Lrow[f] * alpha + rs;
      #pragma unroll
      for (int d = 0; d < 8; d++) O[d][f] *= alpha;
      p0[f] = e0; p1[f] = e1;
    }
    #pragma unroll
    for (int f = 0; f < 8; f++) {
      int m = (lane < 16) ? f : 8 + f;
      P[m * 32 + l16] = __float2bfloat16(p0[f]);
      P[m * 32 + 16 + l16] = __float2bfloat16(p1[f]);
    }
    __builtin_amdgcn_wave_barrier();
    const bht* pr = P + l16 * 32;
    bfrag_t pf = make_frag(pr + kaddA, pr + kaddA + 16);
    #pragma unroll
    for (int d = 0; d < 8; d++) {
      const bht* vr = VT + ((size_t)head * HD_ + d * 16 + l16) * L2_ + kvB + koffB;
      bfrag_t vf = make_frag(vr, vr + 16);
      O[d] = wmma_bf16(pf, vf, O[d]);
    }
    __builtin_amdgcn_wave_barrier();
  }
  #pragma unroll
  for (int f = 0; f < 8; f++) {
    float inv = 1.f / Lrow[f];
    #pragma unroll
    for (int d = 0; d < 8; d++) {
      int col = head * HD_ + d * 16 + l16;
      out[(size_t)qg[f] * N2 + col] = __float2bfloat16(O[d][f] * inv);
    }
  }
}

// ---------- gather gelu(mlp) rows (skip ref rows) into lin2 input cols [H, H+MLPD) ----------
__global__ __launch_bounds__(256) void concat_mlp(const bht* __restrict__ mlp,
                                                  bht* __restrict__ l2in) {
  size_t idx = (size_t)blockIdx.x * 256 + threadIdx.x;  // over LQ * (MLPD/8)
  size_t per = MLPD / 8;
  if (idx >= (size_t)LQ * per) return;
  int q = (int)(idx / per);
  int j = (int)(idx - (size_t)q * per) * 8;
  int src = (q < IMG_) ? q : q + REF_;
  *(uint4*)(l2in + (size_t)q * N2 + H_ + j) =
      *(const uint4*)(mlp + (size_t)src * MLPD + j);
}

extern "C" void kernel_launch(void* const* d_in, const int* in_sizes, int n_in,
                              void* d_out, int out_size, void* d_ws, size_t ws_size,
                              hipStream_t stream) {
  const float* x      = (const float*)d_in[0];
  const float* vec    = (const float*)d_in[1];
  const float* ref    = (const float*)d_in[2];
  const float* cosb   = (const float*)d_in[3];
  const float* sinb   = (const float*)d_in[4];
  const float* mod_w  = (const float*)d_in[5];
  const float* mod_b  = (const float*)d_in[6];
  const float* rf1_w  = (const float*)d_in[7];
  const float* rf1_b  = (const float*)d_in[8];
  const float* rf2_w  = (const float*)d_in[9];
  const float* rf2_b  = (const float*)d_in[10];
  const float* rln_g  = (const float*)d_in[11];
  const float* rln_b  = (const float*)d_in[12];
  const float* lin1_w = (const float*)d_in[13];
  const float* lin1_b = (const float*)d_in[14];
  const float* q_g    = (const float*)d_in[15];
  const float* k_g    = (const float*)d_in[16];
  const float* lin2_w = (const float*)d_in[17];
  const float* lin2_b = (const float*)d_in[18];
  const int*   tmask  = (const int*)d_in[19];
  float* out = (float*)d_out;

  char* ws = (char*)d_ws;
  size_t off = 0;
  auto alloc = [&](size_t bytes) -> char* {
    char* p = ws + off;
    off += (bytes + 255) & ~(size_t)255;
    return p;
  };
  float* modv = (float*)alloc((size_t)3 * H_ * 4);
  bht* refb  = (bht*)alloc((size_t)REF_ * H_ * 2);
  bht* rf1T  = (bht*)alloc((size_t)H_ * H_ * 2);
  bht* rf2T  = (bht*)alloc((size_t)H_ * H_ * 2);
  bht* lin1T = (bht*)alloc((size_t)N1 * H_ * 2);
  bht* lin2T = (bht*)alloc((size_t)H_ * N2 * 2);
  bht* t1    = (bht*)alloc((size_t)REF_ * H_ * 2);
  float* rout = (float*)alloc((size_t)REF_ * H_ * 4);
  bht* xc    = (bht*)alloc((size_t)L2_ * H_ * 2);
  float* qkv = (float*)alloc((size_t)L2_ * 3 * H_ * 4);
  bht* mlp   = (bht*)alloc((size_t)L2_ * MLPD * 2);
  bht* Qb    = (bht*)alloc((size_t)NH_ * LQ * HD_ * 2);
  bht* Kb    = (bht*)alloc((size_t)NH_ * L2_ * HD_ * 2);
  bht* VT    = (bht*)alloc((size_t)NH_ * HD_ * L2_ * 2);
  bht* l2in  = (bht*)alloc((size_t)LQ * N2 * 2);

  // weight conversion (reads fp32 weights exactly once)
  transpose_to_bf16<<<dim3(H_ / 32, H_ / 32), 256, 0, stream>>>(rf1_w, rf1T, H_, H_);
  transpose_to_bf16<<<dim3(H_ / 32, H_ / 32), 256, 0, stream>>>(rf2_w, rf2T, H_, H_);
  transpose_to_bf16<<<dim3(N1 / 32, H_ / 32), 256, 0, stream>>>(lin1_w, lin1T, H_, N1);
  transpose_to_bf16<<<dim3(H_ / 32, N2 / 32), 256, 0, stream>>>(lin2_w, lin2T, N2, H_);
  {
    size_t n4 = (size_t)REF_ * H_ / 4;
    f32_to_bf16<<<(unsigned)((n4 + 255) / 256), 256, 0, stream>>>(ref, refb, n4);
  }

  // modulation + LN paths
  mod_gemv<<<(3 * H_) / 256, 256, 0, stream>>>(vec, mod_w, mod_b, modv);
  ln_mod<<<LQ, 256, 0, stream>>>(x, modv, xc);

  // ref branch: two GEMMs + LN
  gemm_wmma<1><<<dim3(H_ / 128, REF_ / 128), 256, 0, stream>>>(
      refb, rf1T, rf1_b, REF_, H_, H_, nullptr, t1, nullptr, nullptr);
  gemm_wmma<0><<<dim3(H_ / 128, REF_ / 128), 256, 0, stream>>>(
      t1, rf2T, rf2_b, REF_, H_, H_, rout, nullptr, nullptr, nullptr);
  ln_ref<<<REF_, 256, 0, stream>>>(rout, rln_g, rln_b, xc);

  // lin1 split: qkv (fp32) + mlp (gelu->bf16)
  gemm_wmma<0><<<dim3((3 * H_) / 128, L2_ / 128), 256, 0, stream>>>(
      xc, lin1T, lin1_b, L2_, 3 * H_, H_, qkv, nullptr, nullptr, nullptr);
  gemm_wmma<1><<<dim3(MLPD / 128, L2_ / 128), 256, 0, stream>>>(
      xc, lin1T + (size_t)(3 * H_) * H_, lin1_b + 3 * H_, L2_, MLPD, H_,
      nullptr, mlp, nullptr, nullptr);

  // attention
  qkv_prep<<<L2_ * NH_, 128, 0, stream>>>(qkv, cosb, sinb, q_g, k_g, Qb, Kb, VT);
  attn_wmma<<<(NH_ * (LQ / 16)) / 8, 256, 0, stream>>>(Qb, Kb, VT, tmask, l2in);
  {
    size_t n = (size_t)LQ * (MLPD / 8);
    concat_mlp<<<(unsigned)((n + 255) / 256), 256, 0, stream>>>(mlp, l2in);
  }

  // lin2 with fused residual + gate epilogue -> d_out
  gemm_wmma<2><<<dim3(H_ / 128, LQ / 128), 256, 0, stream>>>(
      l2in, lin2T, lin2_b, LQ, H_, N2, out, nullptr, modv + 2 * H_, x);
}
